// TransferPn2_45449343926582
// MI455X (gfx1250) — compile-verified, matrix operand order checked by reading, not run on hardware
//
#include <hip/hip_runtime.h>

// ---------------------------------------------------------------------------
// PointNet++ MSG forward for MI455X (gfx1250, wave32, WMMA).
// GEMM core: v_wmma_f32_16x16x32_f16, 128x64 block tile, 8 waves/block,
// LDS staging via global_load_async_to_lds_b128 when available.
// ---------------------------------------------------------------------------

typedef __attribute__((ext_vector_type(16))) _Float16 v16h;
typedef __attribute__((ext_vector_type(8)))  float    v8f;
typedef int v4i __attribute__((vector_size(16)));
typedef __attribute__((address_space(3))) v4i* lds_v4i_ptr;

#define PN_BN_INV 0.9999950000374997f  // 1/sqrt(1+1e-5)

#define TM 128
#define TN 64
#define TK 32

#if defined(__HIP_DEVICE_COMPILE__)
#  if defined(__has_builtin)
#    if __has_builtin(__builtin_amdgcn_global_load_async_to_lds_b128)
#      define HAVE_ASYNC_LDS 1
#    endif
#    if __has_builtin(__builtin_amdgcn_s_wait_asynccnt)
#      define HAVE_WAIT_ASYNC 1
#    endif
#  endif
#endif

// ---------------------------------------------------------------------------
// Fused GEMM + BN + ReLU:  Y[M,Cout] = act((A[M,Cin] @ W[Cout,Cin]^T)+b)
// gamma!=null -> y = (dot+b)*gamma*BN_INV + beta, relu. Else y = dot + b.
// ---------------------------------------------------------------------------
__global__ __launch_bounds__(256)
void gemm_bn_act_wmma(const _Float16* __restrict__ A,
                      const _Float16* __restrict__ W,
                      const float* __restrict__ bias,
                      const float* __restrict__ gamma,
                      const float* __restrict__ beta,
                      _Float16* __restrict__ Yh,
                      float*    __restrict__ Yf,
                      int M, int Cin, int Cout)
{
  __shared__ __align__(16) _Float16 As[TM][TK];
  __shared__ __align__(16) _Float16 Bs[TN][TK];

  const int tid  = threadIdx.x;
  const int wave = tid >> 5;
  const int lane = tid & 31;
  const int hv   = lane >> 4;   // lane half (K-group select per ISA layout)
  const int lm   = lane & 15;   // row (A) / column (B,D) within tile

  const int row0 = blockIdx.x * TM;
  const int col0 = blockIdx.y * TN;

  v8f acc[4] = {};

  for (int k0 = 0; k0 < Cin; k0 += TK) {
#ifdef HAVE_ASYNC_LDS
    const bool asyncOK = ((row0 + TM) <= M) && ((col0 + TN) <= Cout) &&
                         ((k0 + TK) <= Cin) && ((Cin & 7) == 0);
    if (asyncOK) {
      // A tile: 128 rows x 4 chunks of 16B ; 2 chunks per thread
      for (int u = tid; u < TM * 4; u += 256) {
        int r = u >> 2, c = u & 3;
        const _Float16* g = A + (size_t)(row0 + r) * Cin + k0 + c * 8;
        __builtin_amdgcn_global_load_async_to_lds_b128(
            (v4i*)g, (lds_v4i_ptr)&As[r][c * 8], 0, 0);
      }
      // W tile: 64 rows x 4 chunks of 16B ; 1 chunk per thread
      for (int u = tid; u < TN * 4; u += 256) {
        int r = u >> 2, c = u & 3;
        const _Float16* g = W + (size_t)(col0 + r) * Cin + k0 + c * 8;
        __builtin_amdgcn_global_load_async_to_lds_b128(
            (v4i*)g, (lds_v4i_ptr)&Bs[r][c * 8], 0, 0);
      }
#ifdef HAVE_WAIT_ASYNC
      __builtin_amdgcn_s_wait_asynccnt(0);
#else
      asm volatile("s_wait_asynccnt 0" ::: "memory");
#endif
    } else
#endif
    {
      for (int u = tid; u < TM * TK; u += 256) {
        int r = u / TK, c = u % TK;
        int gr = row0 + r, gc = k0 + c;
        As[r][c] = (gr < M && gc < Cin) ? A[(size_t)gr * Cin + gc]
                                        : (_Float16)0.0f;
      }
      for (int u = tid; u < TN * TK; u += 256) {
        int r = u / TK, c = u % TK;
        int gr = col0 + r, gc = k0 + c;
        Bs[r][c] = (gr < Cout && gc < Cin) ? W[(size_t)gr * Cin + gc]
                                           : (_Float16)0.0f;
      }
    }
    __syncthreads();

    // A fragment: 16x32 f16 (ISA 7.12.2): lanes 0-15 rows, VGPR0-3 K0..7,
    // VGPR4-7 K16..23; lanes 16-31 hold K+8.
    v16h afrag;
#pragma unroll
    for (int e = 0; e < 16; ++e) {
      int kk = (e & 7) + ((e >> 3) << 4) + hv * 8;
      afrag[e] = As[wave * 16 + lm][kk];
    }
#pragma unroll
    for (int t = 0; t < 4; ++t) {
      v16h bfrag;
#pragma unroll
      for (int e = 0; e < 16; ++e) {
        int kk = (e & 7) + ((e >> 3) << 4) + hv * 8;
        bfrag[e] = Bs[t * 16 + lm][kk];
      }
      acc[t] = __builtin_amdgcn_wmma_f32_16x16x32_f16(
          false, afrag, false, bfrag, (short)0, acc[t], false, false);
    }
    __syncthreads();
  }

  // Epilogue. D layout: lane lm = column; VGPR r -> row r + 8*hv.
#pragma unroll
  for (int t = 0; t < 4; ++t) {
    int col = col0 + t * 16 + lm;
    if (col >= Cout) continue;
    float alpha = 1.0f;
    float betav = bias ? bias[col] : 0.0f;
    const bool relu = (gamma != nullptr);
    if (relu) {
      float s = gamma[col] * PN_BN_INV;
      betav = betav * s + beta[col];
      alpha = s;
    }
#pragma unroll
    for (int r = 0; r < 8; ++r) {
      int grow = row0 + wave * 16 + r + 8 * hv;
      if (grow >= M) continue;
      float y = acc[t][r] * alpha + betav;
      if (relu) y = fmaxf(y, 0.0f);
      size_t o = (size_t)grow * Cout + col;
      if (Yh) Yh[o] = (_Float16)y;
      if (Yf) Yf[o] = y;
    }
  }
}

// ---------------------------------------------------------------------------
// Support kernels (bandwidth / latency bound stages)
// ---------------------------------------------------------------------------
__global__ void transpose_bcn_kernel(const float* __restrict__ x,
                                     float* __restrict__ y, int C, int N)
{
  size_t b = blockIdx.y;
  int i = blockIdx.x * blockDim.x + threadIdx.x;
  if (i >= N) return;
  for (int c = 0; c < C; ++c)
    y[(b * N + i) * C + c] = x[(b * C + c) * N + i];
}

__global__ void f32_to_f16_kernel(const float* __restrict__ x,
                                  _Float16* __restrict__ y, long n)
{
  long t = (long)blockIdx.x * blockDim.x + threadIdx.x;
  if (t < n) y[t] = (_Float16)x[t];
}

__global__ void fps_kernel(const float* __restrict__ pts,  // [B,N,3]
                           float* __restrict__ dist,       // [B,N] scratch
                           int* __restrict__ idx,          // [B,np]
                           float* __restrict__ new_xyz,    // [B,np,3]
                           int N, int np)
{
  int b = blockIdx.x;
  const float* P = pts + (size_t)b * N * 3;
  float* D = dist + (size_t)b * N;
  __shared__ float sv[256];
  __shared__ int   si[256];
  __shared__ int   s_far;
  int tid = threadIdx.x;
  for (int i = tid; i < N; i += 256) D[i] = 1.0e10f;
  if (tid == 0) s_far = 0;
  __syncthreads();
  for (int it = 0; it < np; ++it) {
    int far = s_far;
    if (tid == 0) {
      idx[b * np + it] = far;
      new_xyz[((size_t)b * np + it) * 3 + 0] = P[far * 3 + 0];
      new_xyz[((size_t)b * np + it) * 3 + 1] = P[far * 3 + 1];
      new_xyz[((size_t)b * np + it) * 3 + 2] = P[far * 3 + 2];
    }
    float cx = P[far * 3 + 0], cy = P[far * 3 + 1], cz = P[far * 3 + 2];
    float best = -1.0f; int bi = 0;
    for (int i = tid; i < N; i += 256) {
      float dx = P[i * 3 + 0] - cx;
      float dy = P[i * 3 + 1] - cy;
      float dz = P[i * 3 + 2] - cz;
      float d = dx * dx + dy * dy + dz * dz;
      float m = fminf(D[i], d);
      D[i] = m;
      if (m > best) { best = m; bi = i; }
    }
    sv[tid] = best; si[tid] = bi;
    __syncthreads();
    for (int s = 128; s > 0; s >>= 1) {
      if (tid < s && sv[tid + s] > sv[tid]) {
        sv[tid] = sv[tid + s]; si[tid] = si[tid + s];
      }
      __syncthreads();
    }
    if (tid == 0) s_far = si[0];
    __syncthreads();
  }
}

// Sequential scan reproduces sorted-ascending-index + pad-with-first semantics.
__global__ void ball_query_kernel(const float* __restrict__ xyz,      // [B,N,3]
                                  const float* __restrict__ new_xyz,  // [B,S,3]
                                  int* __restrict__ gi,               // [B,S,K]
                                  int Bb, int N, int S, int K, float r2)
{
  int q = blockIdx.x * blockDim.x + threadIdx.x;
  if (q >= Bb * S) return;
  int b = q / S;
  const float* P = xyz + (size_t)b * N * 3;
  float cx = new_xyz[(size_t)q * 3 + 0];
  float cy = new_xyz[(size_t)q * 3 + 1];
  float cz = new_xyz[(size_t)q * 3 + 2];
  int* out = gi + (size_t)q * K;
  int cnt = 0, first = -1;
  for (int i = 0; i < N && cnt < K; ++i) {
    float dx = P[i * 3 + 0] - cx;
    float dy = P[i * 3 + 1] - cy;
    float dz = P[i * 3 + 2] - cz;
    float d = dx * dx + dy * dy + dz * dz;
    if (d <= r2) { if (first < 0) first = i; out[cnt++] = i; }
  }
  if (first < 0) first = 0;
  for (int k = cnt; k < K; ++k) out[k] = first;
}

// out[(b,s,k), :] = concat(feat[gi], xyz[gi]-new_xyz)  -> f16 rows of Cf+3
__global__ void group_kernel(const float* __restrict__ xyz,       // [B,N,3]
                             const float* __restrict__ new_xyz,   // [B,S,3]
                             const _Float16* __restrict__ feat,   // [B,N,Cf]
                             int Cf,
                             const int* __restrict__ gi,          // [B,S,K]
                             _Float16* __restrict__ out,          // [B*S*K, Cf+3]
                             int Bb, int N, int S, int K)
{
  long t = (long)blockIdx.x * blockDim.x + threadIdx.x;
  long total = (long)Bb * S * K;
  if (t >= total) return;
  int k = (int)(t % K);
  long q = t / K;
  int b = (int)(q / S);
  int g = gi[q * K + k];
  const float* P = xyz + ((size_t)b * N + g) * 3;
  const _Float16* F = feat + ((size_t)b * N + g) * Cf;
  _Float16* O = out + (size_t)t * (Cf + 3);
  for (int c = 0; c < Cf; ++c) O[c] = F[c];
  O[Cf + 0] = (_Float16)(P[0] - new_xyz[q * 3 + 0]);
  O[Cf + 1] = (_Float16)(P[1] - new_xyz[q * 3 + 1]);
  O[Cf + 2] = (_Float16)(P[2] - new_xyz[q * 3 + 2]);
}

__global__ void maxpool_kernel(const _Float16* __restrict__ x,  // [BS,K,C]
                               _Float16* __restrict__ out,      // [BS,Cstride]
                               long BS, int K, int C, int Cstride, int coff)
{
  long t = (long)blockIdx.x * blockDim.x + threadIdx.x;
  if (t >= BS * (long)C) return;
  int c = (int)(t % C);
  long bs = t / C;
  const _Float16* X = x + bs * (long)K * C + c;
  float m = -3.0e38f;
  for (int k = 0; k < K; ++k) {
    float v = (float)X[(long)k * C];
    m = v > m ? v : m;
  }
  out[bs * (long)Cstride + coff + c] = (_Float16)m;
}

__global__ void concat_xyz_feat_kernel(const float* __restrict__ xyz,   // [rows,3]
                                       const _Float16* __restrict__ f,  // [rows,Cf]
                                       _Float16* __restrict__ out,      // [rows,3+Cf]
                                       long rows, int Cf)
{
  long t = (long)blockIdx.x * blockDim.x + threadIdx.x;
  int C = 3 + Cf;
  if (t >= rows * (long)C) return;
  int c = (int)(t % C);
  long r = t / C;
  out[t] = (c < 3) ? (_Float16)xyz[r * 3 + c] : f[r * (long)Cf + (c - 3)];
}

__global__ void concat_bcast_kernel(const _Float16* __restrict__ a,  // [B,S,Ca]
                                    const _Float16* __restrict__ g,  // [B,Cg]
                                    _Float16* __restrict__ out,      // [B,S,Ca+Cg]
                                    int Bb, int S, int Ca, int Cg)
{
  long t = (long)blockIdx.x * blockDim.x + threadIdx.x;
  int C = Ca + Cg;
  long total = (long)Bb * S * C;
  if (t >= total) return;
  int c = (int)(t % C);
  long bs = t / C;
  int b = (int)(bs / S);
  out[t] = (c < Ca) ? a[bs * (long)Ca + c] : g[(long)b * Cg + (c - Ca)];
}

__global__ void copy_chan_f16_kernel(const _Float16* __restrict__ src,
                                     _Float16* __restrict__ dst,
                                     long rows, int C, int Cstride, int coff)
{
  long t = (long)blockIdx.x * blockDim.x + threadIdx.x;
  if (t >= rows * (long)C) return;
  int c = (int)(t % C);
  long r = t / C;
  dst[r * (long)Cstride + coff + c] = src[r * (long)C + c];
}

// l0_feat = concat(xyz, xyz) -> 6 channels written at stride Cstride
__global__ void l0feat_kernel(const float* __restrict__ pts,
                              _Float16* __restrict__ dst, long rows, int Cstride)
{
  long t = (long)blockIdx.x * blockDim.x + threadIdx.x;
  if (t >= rows) return;
  _Float16* o = dst + t * (long)Cstride;
  _Float16 x = (_Float16)pts[t * 3 + 0];
  _Float16 y = (_Float16)pts[t * 3 + 1];
  _Float16 z = (_Float16)pts[t * 3 + 2];
  o[0] = x; o[1] = y; o[2] = z; o[3] = x; o[4] = y; o[5] = z;
}

// 3-NN inverse-distance interpolation; writes C2 channels at coff.
__global__ void three_interp_kernel(const float* __restrict__ xyz1,      // [B,Nn,3]
                                    const float* __restrict__ xyz2,      // [B,S,3]
                                    const _Float16* __restrict__ feat2,  // [B,S,C2]
                                    _Float16* __restrict__ out,          // [B,Nn,Cstride]
                                    int Bb, int Nn, int S, int C2,
                                    int Cstride, int coff)
{
  long t = (long)blockIdx.x * blockDim.x + threadIdx.x;
  if (t >= (long)Bb * Nn) return;
  int b = (int)(t / Nn);
  const float* p = xyz1 + (size_t)t * 3;
  const float* Q = xyz2 + (size_t)b * S * 3;
  float d0 = 3.0e38f, d1 = 3.0e38f, d2 = 3.0e38f;
  int i0 = 0, i1 = 0, i2 = 0;
  for (int s = 0; s < S; ++s) {
    float dx = Q[s * 3 + 0] - p[0];
    float dy = Q[s * 3 + 1] - p[1];
    float dz = Q[s * 3 + 2] - p[2];
    float d = dx * dx + dy * dy + dz * dz;
    if (d < d0)      { d2 = d1; i2 = i1; d1 = d0; i1 = i0; d0 = d; i0 = s; }
    else if (d < d1) { d2 = d1; i2 = i1; d1 = d; i1 = s; }
    else if (d < d2) { d2 = d; i2 = s; }
  }
  float w0 = 1.0f / (d0 + 1e-8f);
  float w1 = 1.0f / (d1 + 1e-8f);
  float w2 = 1.0f / (d2 + 1e-8f);
  float ws = w0 + w1 + w2;
  w0 /= ws; w1 /= ws; w2 /= ws;
  const _Float16* F = feat2 + (size_t)b * S * C2;
  _Float16* O = out + (size_t)t * Cstride + coff;
  for (int c = 0; c < C2; ++c) {
    float v = w0 * (float)F[(size_t)i0 * C2 + c] +
              w1 * (float)F[(size_t)i1 * C2 + c] +
              w2 * (float)F[(size_t)i2 * C2 + c];
    O[c] = (_Float16)v;
  }
}

// ---------------------------------------------------------------------------
// Host orchestration
// ---------------------------------------------------------------------------
struct Lay {
  const float* W; const float* b; const float* g; const float* be;
  int cin; int cout;
};

static void run_mlp(const Lay* Ls, int nl, const _Float16* X0, long M,
                    _Float16* pA, _Float16* pB, _Float16* Wh,
                    _Float16* finalH, float* finalF,
                    hipStream_t stream, const _Float16** outp)
{
  _Float16* pp[2] = {pA, pB};
  int which = (X0 == (const _Float16*)pA) ? 1 : 0;
  const _Float16* cur = X0;
  for (int i = 0; i < nl; ++i) {
    const bool last = (i == nl - 1);
    _Float16* dh; float* df = nullptr;
    if (last && (finalH != nullptr || finalF != nullptr)) {
      dh = finalH; df = finalF;
    } else {
      dh = pp[which]; which ^= 1;
    }
    long wn = (long)Ls[i].cout * Ls[i].cin;
    f32_to_f16_kernel<<<(unsigned)((wn + 255) / 256), 256, 0, stream>>>(
        Ls[i].W, Wh, wn);
    dim3 grid((unsigned)((M + TM - 1) / TM),
              (unsigned)((Ls[i].cout + TN - 1) / TN));
    gemm_bn_act_wmma<<<grid, 256, 0, stream>>>(
        cur, Wh, Ls[i].b, Ls[i].g, Ls[i].be, dh, df,
        (int)M, Ls[i].cin, Ls[i].cout);
    if (dh) cur = dh;
  }
  if (outp) *outp = cur;
}

extern "C" void kernel_launch(void* const* d_in, const int* in_sizes, int n_in,
                              void* d_out, int out_size, void* d_ws, size_t ws_size,
                              hipStream_t stream)
{
  (void)in_sizes; (void)n_in; (void)out_size; (void)ws_size;
  const int B = 8, N = 8192, NP1 = 512, NP2 = 128;

  const float* xyz = (const float*)d_in[0];
  int pi = 1;
  auto TL = [&](int cin, int cout) -> Lay {
    Lay L;
    L.W  = (const float*)d_in[pi++];
    L.b  = (const float*)d_in[pi++];
    L.g  = (const float*)d_in[pi++];
    L.be = (const float*)d_in[pi++];
    L.cin = cin; L.cout = cout;
    return L;
  };
  Lay sa1a[3] = {TL(6, 32),   TL(32, 32),  TL(32, 64)};
  Lay sa1b[3] = {TL(6, 64),   TL(64, 64),  TL(64, 128)};
  Lay sa1c[3] = {TL(6, 64),   TL(64, 96),  TL(96, 128)};
  Lay sa2a[3] = {TL(323, 128), TL(128, 128), TL(128, 256)};
  Lay sa2b[3] = {TL(323, 128), TL(128, 196), TL(196, 256)};
  Lay sa3l[3] = {TL(515, 256), TL(256, 512), TL(512, 1024)};
  Lay fp3l[2] = {TL(1536, 256), TL(256, 256)};
  Lay fp2l[2] = {TL(576, 256),  TL(256, 128)};
  Lay fp1l[2] = {TL(134, 128),  TL(128, 128)};
  Lay head1[1] = {TL(128, 128)};
  Lay conv2l[1];
  conv2l[0].W = (const float*)d_in[pi++];
  conv2l[0].b = (const float*)d_in[pi++];
  conv2l[0].g = nullptr; conv2l[0].be = nullptr;
  conv2l[0].cin = 128; conv2l[0].cout = 50;

  // ---- workspace bump allocator ----
  char* wsp = (char*)d_ws;
  auto bump = [&](size_t bytes) -> void* {
    void* r = (void*)wsp;
    wsp += (bytes + 255) & ~(size_t)255;
    return r;
  };
  float*    pts    = (float*)   bump((size_t)B * N * 3 * 4);
  _Float16* pts_h  = (_Float16*)bump((size_t)B * N * 3 * 2);
  float*    fdist  = (float*)   bump((size_t)B * N * 4);
  int*      idx1   = (int*)     bump((size_t)B * NP1 * 4);
  float*    l1xyz  = (float*)   bump((size_t)B * NP1 * 3 * 4);
  int*      idx2   = (int*)     bump((size_t)B * NP2 * 4);
  float*    l2xyz  = (float*)   bump((size_t)B * NP2 * 3 * 4);
  int*      gi     = (int*)     bump((size_t)B * NP1 * 128 * 4);
  _Float16* Wh     = (_Float16*)bump((size_t)1024 * 1536 * 2);
  _Float16* l1pts  = (_Float16*)bump((size_t)B * NP1 * 320 * 2);
  _Float16* l2pts  = (_Float16*)bump((size_t)B * NP2 * 512 * 2);
  _Float16* l3pts  = (_Float16*)bump((size_t)B * 1024 * 2);
  _Float16* l2fp   = (_Float16*)bump((size_t)B * NP2 * 256 * 2);
  _Float16* l1fp   = (_Float16*)bump((size_t)B * NP1 * 128 * 2);
  _Float16* l0x    = (_Float16*)bump((size_t)B * N * 134 * 2);
  size_t bigElems  = (size_t)B * NP1 * 128 * 128;  // largest M*C in network
  _Float16* bufA   = (_Float16*)bump(bigElems * 2);
  _Float16* bufB   = (_Float16*)bump(bigElems * 2);

  // ---- l0: transpose (B,3,N)->(B,N,3), f16 copy of coords ----
  transpose_bcn_kernel<<<dim3((N + 255) / 256, B), 256, 0, stream>>>(xyz, pts, 3, N);
  f32_to_f16_kernel<<<(unsigned)(((size_t)B * N * 3 + 255) / 256), 256, 0, stream>>>(
      pts, pts_h, (long)B * N * 3);

  const _Float16* co = nullptr;

  // ---- SA1 (MSG, 3 branches) ----
  fps_kernel<<<B, 256, 0, stream>>>(pts, fdist, idx1, l1xyz, N, NP1);
  {
    int   Ks[3] = {32, 64, 128};
    float Rs[3] = {0.1f, 0.2f, 0.4f};
    const Lay* Br[3] = {sa1a, sa1b, sa1c};
    int coff = 0;
    for (int br = 0; br < 3; ++br) {
      int K = Ks[br]; float r2 = Rs[br] * Rs[br];
      long M = (long)B * NP1 * K;
      ball_query_kernel<<<(unsigned)((B * NP1 + 255) / 256), 256, 0, stream>>>(
          pts, l1xyz, gi, B, N, NP1, K, r2);
      group_kernel<<<(unsigned)((M + 255) / 256), 256, 0, stream>>>(
          pts, l1xyz, pts_h, 3, gi, bufA, B, N, NP1, K);
      run_mlp(Br[br], 3, bufA, M, bufA, bufB, Wh, nullptr, nullptr, stream, &co);
      int cout = Br[br][2].cout;
      long BS = (long)B * NP1;
      maxpool_kernel<<<(unsigned)((BS * cout + 255) / 256), 256, 0, stream>>>(
          co, l1pts, BS, K, cout, 320, coff);
      coff += cout;
    }
  }

  // ---- SA2 (MSG, 2 branches) ----
  fps_kernel<<<B, 256, 0, stream>>>(l1xyz, fdist, idx2, l2xyz, NP1, NP2);
  {
    int   Ks[2] = {64, 128};
    float Rs[2] = {0.4f, 0.8f};
    const Lay* Br[2] = {sa2a, sa2b};
    int coff = 0;
    for (int br = 0; br < 2; ++br) {
      int K = Ks[br]; float r2 = Rs[br] * Rs[br];
      long M = (long)B * NP2 * K;
      ball_query_kernel<<<(unsigned)((B * NP2 + 255) / 256), 256, 0, stream>>>(
          l1xyz, l2xyz, gi, B, NP1, NP2, K, r2);
      group_kernel<<<(unsigned)((M + 255) / 256), 256, 0, stream>>>(
          l1xyz, l2xyz, l1pts, 320, gi, bufA, B, NP1, NP2, K);
      run_mlp(Br[br], 3, bufA, M, bufA, bufB, Wh, nullptr, nullptr, stream, &co);
      long BS = (long)B * NP2;
      maxpool_kernel<<<(unsigned)((BS * 256 + 255) / 256), 256, 0, stream>>>(
          co, l2pts, BS, K, 256, 512, coff);
      coff += 256;
    }
  }

  // ---- SA3 (group-all) ----
  long r3 = (long)B * NP2;
  concat_xyz_feat_kernel<<<(unsigned)((r3 * 515 + 255) / 256), 256, 0, stream>>>(
      l2xyz, l2pts, bufA, r3, 512);
  run_mlp(sa3l, 3, bufA, r3, bufA, bufB, Wh, nullptr, nullptr, stream, &co);
  maxpool_kernel<<<(unsigned)(((long)B * 1024 + 255) / 256), 256, 0, stream>>>(
      co, l3pts, (long)B, NP2, 1024, 1024, 0);

  // ---- FP3 (S==1 broadcast) ----
  concat_bcast_kernel<<<(unsigned)((r3 * 1536 + 255) / 256), 256, 0, stream>>>(
      l2pts, l3pts, bufA, B, NP2, 512, 1024);
  run_mlp(fp3l, 2, bufA, r3, bufA, bufB, Wh, l2fp, nullptr, stream, nullptr);

  // ---- FP2 (3-NN l2 -> l1) ----
  long r2rows = (long)B * NP1;
  copy_chan_f16_kernel<<<(unsigned)((r2rows * 320 + 255) / 256), 256, 0, stream>>>(
      l1pts, bufA, r2rows, 320, 576, 0);
  three_interp_kernel<<<(unsigned)((r2rows + 255) / 256), 256, 0, stream>>>(
      l1xyz, l2xyz, l2fp, bufA, B, NP1, NP2, 256, 576, 320);
  run_mlp(fp2l, 2, bufA, r2rows, bufA, bufB, Wh, l1fp, nullptr, stream, nullptr);

  // ---- FP1 (3-NN l1 -> l0) ----
  long r1 = (long)B * N;
  l0feat_kernel<<<(unsigned)((r1 + 255) / 256), 256, 0, stream>>>(pts, l0x, r1, 134);
  three_interp_kernel<<<(unsigned)((r1 + 255) / 256), 256, 0, stream>>>(
      pts, l1xyz, l1fp, l0x, B, N, NP1, 128, 134, 6);
  run_mlp(fp1l, 2, l0x, r1, bufA, bufB, Wh, nullptr, nullptr, stream, &co);

  // ---- head + final linear (fp32 out) ----
  run_mlp(head1, 1, co, r1, bufA, bufB, Wh, nullptr, nullptr, stream, &co);
  run_mlp(conv2l, 1, co, r1, bufA, bufB, Wh, nullptr, (float*)d_out, stream, nullptr);
}
